// MambaBlock_29910152249691
// MI455X (gfx1250) — matrix operand order repaired and checked by reading, not compiled, not run on hardware
//
#include <hip/hip_runtime.h>
#include <hip/hip_bf16.h>
#include <math.h>

// ---- problem dims (match reference) ----
#define D_MODEL 1024
#define ED      2048
#define LSEQ    2048
#define RNK     64
#define NST     16
#define KCONV   4
#define NPROJ   96        // R + 2N
#define NPAD    128       // padded to multiple of the WMMA block tile
#define CHUNK   128       // scan chunk length
#define NCH     (LSEQ / CHUNK)   // 16 chunks

typedef __bf16 bf16;
typedef __attribute__((ext_vector_type(16))) __bf16 bf16x16;
typedef __attribute__((ext_vector_type(8)))  __bf16 bf16x8;
typedef __attribute__((ext_vector_type(8)))  float  f32x8;

__device__ __forceinline__ float sigmoidf_(float v) { return 1.0f / (1.0f + __expf(-v)); }

// ---------------------------------------------------------------------------
// f32 -> bf16 cast with optional zero padding (n_src <= n_total)
// ---------------------------------------------------------------------------
__global__ void cast_bf16_pad(const float* __restrict__ src, bf16* __restrict__ dst,
                              int n_src, int n_total) {
  int i = blockIdx.x * blockDim.x + threadIdx.x;
  if (i < n_total) dst[i] = (i < n_src) ? (bf16)src[i] : (bf16)0.0f;
}

// ---------------------------------------------------------------------------
// C[M,N] = A[M,K] * B[N,K]^T   (A,B row-major bf16, K contiguous; C f32)
// block = 128 threads (4 waves) in 2x2; wave tile 64x64 (4x4 WMMA accums).
// 16 WMMAs per 16 b128 loads -> ~32 flops/byte from cache.
// mode 0: C = acc;  mode 1: C = softplus(acc + bias[n])
// ---------------------------------------------------------------------------
__global__ void gemm_bf16_wmma(const bf16* __restrict__ A, int lda,
                               const bf16* __restrict__ B, int ldb,
                               float* __restrict__ C, int ldc,
                               int K, const float* __restrict__ bias, int mode) {
  const int lane = threadIdx.x & 31;
  const int wave = threadIdx.x >> 5;
  const int m16  = lane & 15;
  const int half = lane >> 4;           // 0 or 1
  const int row0 = blockIdx.y * 128 + (wave >> 1) * 64;
  const int col0 = blockIdx.x * 128 + (wave & 1) * 64;

  f32x8 acc[4][4];
#pragma unroll
  for (int i = 0; i < 4; ++i)
#pragma unroll
    for (int j = 0; j < 4; ++j) acc[i][j] = (f32x8){};

  // A 16x32 frag: lane m holds K=[8h..8h+7] and [16+8h..16+8h+7]
  // B 32x16 frag: lane n holds K=[8h..8h+7] and [16+8h..16+8h+7] of W row n
  const bf16* ap[4];
  const bf16* bp[4];
#pragma unroll
  for (int i = 0; i < 4; ++i) {
    ap[i] = A + (size_t)(row0 + 16 * i + m16) * lda + 8 * half;
    bp[i] = B + (size_t)(col0 + 16 * i + m16) * ldb + 8 * half;
  }

  for (int k0 = 0; k0 < K; k0 += 32) {
    if (k0 + 32 < K) {   // near-scope CDNA5 prefetch of the next K slice
#pragma unroll
      for (int i = 0; i < 4; ++i) {
        __builtin_prefetch(ap[i] + k0 + 32, 0, 3);
        __builtin_prefetch(bp[i] + k0 + 32, 0, 3);
      }
    }
    bf16x16 af[4], bfr[4];
#pragma unroll
    for (int i = 0; i < 4; ++i) {
      bf16x8 lo = *reinterpret_cast<const bf16x8*>(ap[i] + k0);
      bf16x8 hi = *reinterpret_cast<const bf16x8*>(ap[i] + k0 + 16);
#pragma unroll
      for (int q = 0; q < 8; ++q) { af[i][q] = lo[q]; af[i][8 + q] = hi[q]; }
    }
#pragma unroll
    for (int j = 0; j < 4; ++j) {
      bf16x8 lo = *reinterpret_cast<const bf16x8*>(bp[j] + k0);
      bf16x8 hi = *reinterpret_cast<const bf16x8*>(bp[j] + k0 + 16);
#pragma unroll
      for (int q = 0; q < 8; ++q) { bfr[j][q] = lo[q]; bfr[j][8 + q] = hi[q]; }
    }
#pragma unroll
    for (int i = 0; i < 4; ++i)
#pragma unroll
      for (int j = 0; j < 4; ++j)
        acc[i][j] = __builtin_amdgcn_wmma_f32_16x16x32_bf16(
            false, af[i], false, bfr[j], (short)0, acc[i][j], false, false);
  }

  // C/D f32 16x16: VGPR r -> row r (lanes 0-15) / row 8+r (lanes 16-31)
#pragma unroll
  for (int i = 0; i < 4; ++i) {
#pragma unroll
    for (int j = 0; j < 4; ++j) {
      int n = col0 + 16 * j + m16;
      float bval = (mode == 1) ? bias[n] : 0.0f;
#pragma unroll
      for (int r = 0; r < 8; ++r) {
        float v = acc[i][j][r];
        if (mode == 1) {
          float s = v + bval;
          v = (s > 20.0f) ? s : log1pf(__expf(s));   // softplus
        }
        C[(size_t)(row0 + 16 * i + r + 8 * half) * ldc + n] = v;
      }
    }
  }
}

// ---------------------------------------------------------------------------
// Fused causal depthwise conv (K=4) + bias + SiLU; emits f32 and bf16 copies.
// ---------------------------------------------------------------------------
__global__ void conv_silu_kernel(const float* __restrict__ xz,
                                 const float* __restrict__ conv_w,
                                 const float* __restrict__ conv_b,
                                 float* __restrict__ xr, bf16* __restrict__ xrb) {
  int idx = blockIdx.x * blockDim.x + threadIdx.x;
  if (idx >= LSEQ * ED) return;
  int e = idx & (ED - 1);
  int l = idx >> 11;                     // ED == 2048
  float acc = conv_b[e];
#pragma unroll
  for (int j = 0; j < KCONV; ++j) {
    int t = l - (KCONV - 1) + j;
    if (t >= 0) acc += conv_w[e * KCONV + j] * xz[(size_t)t * (2 * ED) + e];
  }
  float s = acc * sigmoidf_(acc);
  xr[idx]  = s;
  xrb[idx] = (bf16)s;
}

// ---------------------------------------------------------------------------
// Chunked selective scan, pass A: per (chunk c, channel e) compute with h0=0:
//   P[n] = prod_t dA,  H[n] = partial state at chunk end.
// 32768 threads -> 16x the parallelism of the naive per-channel scan.
// ---------------------------------------------------------------------------
__global__ void scan_chunk_partial(const float* __restrict__ delta,
                                   const float* __restrict__ xr,
                                   const float* __restrict__ dbc,
                                   const float* __restrict__ A_log,
                                   float* __restrict__ chP, float* __restrict__ chH) {
  int idx = blockIdx.x * blockDim.x + threadIdx.x;   // [0, ED*NCH)
  if (idx >= ED * NCH) return;
  int e = idx & (ED - 1);
  int c = idx >> 11;
  float a[NST], h[NST], P[NST];
#pragma unroll
  for (int n = 0; n < NST; ++n) {
    a[n] = -__expf(A_log[e * NST + n]);
    h[n] = 0.0f; P[n] = 1.0f;
  }
  int t0 = c * CHUNK;
  for (int t = t0; t < t0 + CHUNK; ++t) {
    float dlt = delta[(size_t)t * ED + e];
    float xv  = xr[(size_t)t * ED + e];
    const float* bc = dbc + (size_t)t * NPAD;
    float dx = dlt * xv;
#pragma unroll
    for (int n = 0; n < NST; ++n) {
      float da = __expf(dlt * a[n]);
      P[n] *= da;
      h[n] = da * h[n] + bc[64 + n] * dx;
    }
  }
  size_t base = (size_t)idx * NST;
#pragma unroll
  for (int n = 0; n < NST; ++n) { chP[base + n] = P[n]; chH[base + n] = h[n]; }
}

// ---------------------------------------------------------------------------
// Pass B: tiny cross-chunk serial combine (16 steps per channel):
//   hInit[c] = state entering chunk c;  h <- P_c * h + H_c
// ---------------------------------------------------------------------------
__global__ void scan_chunk_combine(const float* __restrict__ chP,
                                   const float* __restrict__ chH,
                                   float* __restrict__ hInit) {
  int e = blockIdx.x * blockDim.x + threadIdx.x;
  if (e >= ED) return;
  float hs[NST];
#pragma unroll
  for (int n = 0; n < NST; ++n) hs[n] = 0.0f;
  for (int c = 0; c < NCH; ++c) {
    size_t base = ((size_t)c * ED + e) * NST;
#pragma unroll
    for (int n = 0; n < NST; ++n) {
      hInit[base + n] = hs[n];
      hs[n] = chP[base + n] * hs[n] + chH[base + n];
    }
  }
}

// ---------------------------------------------------------------------------
// Pass C: replay each chunk from its correct initial state and emit y.
// ---------------------------------------------------------------------------
__global__ void scan_chunk_emit(const float* __restrict__ delta,
                                const float* __restrict__ xr,
                                const float* __restrict__ dbc,
                                const float* __restrict__ A_log,
                                const float* __restrict__ Dp,
                                const float* __restrict__ hInit,
                                float* __restrict__ y) {
  int idx = blockIdx.x * blockDim.x + threadIdx.x;   // [0, ED*NCH)
  if (idx >= ED * NCH) return;
  int e = idx & (ED - 1);
  int c = idx >> 11;
  float a[NST], h[NST];
  size_t base = (size_t)idx * NST;
#pragma unroll
  for (int n = 0; n < NST; ++n) {
    a[n] = -__expf(A_log[e * NST + n]);
    h[n] = hInit[base + n];
  }
  float dd = Dp[e];
  int t0 = c * CHUNK;
  for (int t = t0; t < t0 + CHUNK; ++t) {
    float dlt = delta[(size_t)t * ED + e];
    float xv  = xr[(size_t)t * ED + e];
    const float* bc = dbc + (size_t)t * NPAD;
    float dx = dlt * xv;
    float yacc = 0.0f;
#pragma unroll
    for (int n = 0; n < NST; ++n) {
      float da = __expf(dlt * a[n]);
      h[n] = da * h[n] + bc[64 + n] * dx;
      yacc += h[n] * bc[80 + n];
    }
    y[(size_t)t * ED + e] = yacc + dd * xv;
  }
}

// ---------------------------------------------------------------------------
// yb = bf16( y * silu(z) ),  z = xz[:, ED:2*ED]
// ---------------------------------------------------------------------------
__global__ void gate_cast_kernel(const float* __restrict__ y,
                                 const float* __restrict__ xz,
                                 bf16* __restrict__ yb) {
  int idx = blockIdx.x * blockDim.x + threadIdx.x;
  if (idx >= LSEQ * ED) return;
  int e = idx & (ED - 1);
  int l = idx >> 11;
  float z = xz[(size_t)l * (2 * ED) + ED + e];
  yb[idx] = (bf16)(y[idx] * (z * sigmoidf_(z)));
}

// ---------------------------------------------------------------------------
extern "C" void kernel_launch(void* const* d_in, const int* in_sizes, int n_in,
                              void* d_out, int out_size, void* d_ws, size_t ws_size,
                              hipStream_t stream) {
  const float* x         = (const float*)d_in[0];   // [1, L, D_MODEL]
  const float* in_proj_w = (const float*)d_in[1];   // [2*ED, D_MODEL]
  const float* conv_w    = (const float*)d_in[2];   // [ED, 1, K]
  const float* conv_b    = (const float*)d_in[3];   // [ED]
  const float* x_proj_w  = (const float*)d_in[4];   // [R+2N, ED]
  const float* dt_proj_w = (const float*)d_in[5];   // [ED, R]
  const float* dt_proj_b = (const float*)d_in[6];   // [ED]
  const float* A_log     = (const float*)d_in[7];   // [ED, N]
  const float* Dp        = (const float*)d_in[8];   // [ED]
  const float* out_proj_w= (const float*)d_in[9];   // [D_MODEL, ED]
  float* out             = (float*)d_out;           // [1, L, D_MODEL] f32

  char* ws = (char*)d_ws;
  size_t off = 0;
  auto alloc = [&](size_t bytes) -> char* {
    char* p = ws + off;
    off += (bytes + 255) & ~size_t(255);
    return p;
  };

  bf16*  xb    = (bf16*) alloc((size_t)LSEQ * D_MODEL * 2);
  bf16*  wbin  = (bf16*) alloc((size_t)2 * ED * D_MODEL * 2);
  float* xz    = (float*)alloc((size_t)LSEQ * 2 * ED * 4);
  float* xr    = (float*)alloc((size_t)LSEQ * ED * 4);
  bf16*  xrb   = (bf16*) alloc((size_t)LSEQ * ED * 2);
  bf16*  wxp   = (bf16*) alloc((size_t)NPAD * ED * 2);
  float* dbc   = (float*)alloc((size_t)LSEQ * NPAD * 4);     // [delta_r | B | C | 0]
  bf16*  dbcb  = (bf16*) alloc((size_t)LSEQ * NPAD * 2);
  bf16*  wdt   = (bf16*) alloc((size_t)ED * RNK * 2);
  float* delta = (float*)alloc((size_t)LSEQ * ED * 4);
  float* ybuf  = (float*)alloc((size_t)LSEQ * ED * 4);
  bf16*  yb    = (bf16*) alloc((size_t)LSEQ * ED * 2);
  bf16*  wout  = (bf16*) alloc((size_t)D_MODEL * ED * 2);
  float* chP   = (float*)alloc((size_t)NCH * ED * NST * 4);  // chunk decay products
  float* chH   = (float*)alloc((size_t)NCH * ED * NST * 4);  // chunk partial states
  float* hInit = (float*)alloc((size_t)NCH * ED * NST * 4);  // per-chunk initial states

  const int TB = 256;
  auto nb = [](int n, int b) { return (n + b - 1) / b; };

  // 1) weight / input casts
  cast_bf16_pad<<<nb(LSEQ * D_MODEL, TB), TB, 0, stream>>>(x, xb, LSEQ * D_MODEL, LSEQ * D_MODEL);
  cast_bf16_pad<<<nb(2 * ED * D_MODEL, TB), TB, 0, stream>>>(in_proj_w, wbin, 2 * ED * D_MODEL, 2 * ED * D_MODEL);
  cast_bf16_pad<<<nb(NPAD * ED, TB), TB, 0, stream>>>(x_proj_w, wxp, NPROJ * ED, NPAD * ED);
  cast_bf16_pad<<<nb(ED * RNK, TB), TB, 0, stream>>>(dt_proj_w, wdt, ED * RNK, ED * RNK);
  cast_bf16_pad<<<nb(D_MODEL * ED, TB), TB, 0, stream>>>(out_proj_w, wout, D_MODEL * ED, D_MODEL * ED);

  // 2) in_proj:  xz[L, 2*ED] = x[L, D] * in_proj_w[2*ED, D]^T
  gemm_bf16_wmma<<<dim3((2 * ED) / 128, LSEQ / 128), 128, 0, stream>>>(
      xb, D_MODEL, wbin, D_MODEL, xz, 2 * ED, D_MODEL, nullptr, 0);

  // 3) causal depthwise conv + SiLU
  conv_silu_kernel<<<nb(LSEQ * ED, TB), TB, 0, stream>>>(xz, conv_w, conv_b, xr, xrb);

  // 4) x_proj:  dbc[L, 128] = xr[L, ED] * x_proj_w_pad[128, ED]^T
  gemm_bf16_wmma<<<dim3(NPAD / 128, LSEQ / 128), 128, 0, stream>>>(
      xrb, ED, wxp, ED, dbc, NPAD, ED, nullptr, 0);
  cast_bf16_pad<<<nb(LSEQ * NPAD, TB), TB, 0, stream>>>(dbc, dbcb, LSEQ * NPAD, LSEQ * NPAD);

  // 5) dt_proj + softplus:  delta[L, ED] = softplus(delta_r[L, R] * dt_proj_w[ED, R]^T + b)
  gemm_bf16_wmma<<<dim3(ED / 128, LSEQ / 128), 128, 0, stream>>>(
      dbcb, NPAD, wdt, RNK, delta, ED, RNK, dt_proj_b, 1);

  // 6) chunked selective scan (partial -> combine -> emit)
  scan_chunk_partial<<<nb(ED * NCH, TB), TB, 0, stream>>>(delta, xr, dbc, A_log, chP, chH);
  scan_chunk_combine<<<ED / TB, TB, 0, stream>>>(chP, chH, hInit);
  scan_chunk_emit<<<nb(ED * NCH, TB), TB, 0, stream>>>(delta, xr, dbc, A_log, Dp, hInit, ybuf);

  // 7) gate with silu(z), cast to bf16
  gate_cast_kernel<<<nb(LSEQ * ED, TB), TB, 0, stream>>>(ybuf, xz, yb);

  // 8) out_proj:  out[L, D] = yb[L, ED] * out_proj_w[D, ED]^T
  gemm_bf16_wmma<<<dim3(D_MODEL / 128, LSEQ / 128), 128, 0, stream>>>(
      yb, ED, wout, ED, out, D_MODEL, ED, nullptr, 0);
}
